// ANEDecoder_9191230013800
// MI455X (gfx1250) — matrix operand order repaired and checked by reading, not compiled, not run on hardware
//
#include <hip/hip_runtime.h>
#include <hip/hip_bf16.h>
#include <math.h>

typedef __attribute__((ext_vector_type(16))) __bf16 v16bf;
typedef __attribute__((ext_vector_type(8)))  float  v8f;

#define B_   2
#define D_   2048
#define S_   128
#define A_   1536
#define T_   512
#define HQ_  16
#define HKV_ 4
#define G_   4
#define HD_  128
#define HC_  16
#define HDC_ 128
#define FF_  8192
#define L_   2
#define NEG_ (-30000.0f)
#define EPS_ 1e-5f
#define SCALE_ 0.08838834764831845f   // 1/sqrt(128)

// ---------- helpers ----------

__device__ __forceinline__ unsigned int pack2bf(float a, float b) {
  unsigned int ua = __float_as_uint(a);
  unsigned int ub = __float_as_uint(b);
  ua = (ua + 0x7FFFu + ((ua >> 16) & 1u)) >> 16;   // RNE truncate to bf16
  ub = (ub + 0x7FFFu + ((ub >> 16) & 1u)) >> 16;
  return (ua & 0xFFFFu) | (ub << 16);
}

union FragU { unsigned int u[8]; v16bf v; };

// A-matrix 16x32 bf16 frag: lane l -> row (passed), K = hi*16..hi*16+15 (contiguous)
__device__ __forceinline__ v16bf load_frag_a(const unsigned int* lds, int row, int lane) {
  FragU f;
  const unsigned int* p = lds + row * 16;
  const int hi = (lane >> 4) * 8;
#pragma unroll
  for (int i = 0; i < 4; ++i) f.u[i] = p[hi + i];
#pragma unroll
  for (int i = 0; i < 4; ++i) f.u[4 + i] = p[hi + 4 + i];
  return f.v;
}

// B-matrix 32x16 bf16 frag: lane l -> col (passed), K = hi*16..hi*16+15 (contiguous)
__device__ __forceinline__ v16bf load_frag_b(const unsigned int* lds, int col, int lane) {
  FragU f;
  const unsigned int* p = lds + col * 16 + (lane >> 4) * 8;
#pragma unroll
  for (int i = 0; i < 8; ++i) f.u[i] = p[i];
  return f.v;
}

// ---------- RMSNorm: x (B,D,S) column-wise over D ----------
__global__ __launch_bounds__(256)
void rmsnorm_kernel(const float* __restrict__ X, const float* __restrict__ Wn,
                    float* __restrict__ OUT) {
  const int b = blockIdx.x / S_;
  const int s = blockIdx.x % S_;
  const float* xb = X + (size_t)b * D_ * S_ + s;
  __shared__ float red[256];
  const int t = threadIdx.x;
  float sum = 0.f;
  for (int d = t; d < D_; d += 256) { float v = xb[(size_t)d * S_]; sum += v * v; }
  red[t] = sum; __syncthreads();
  for (int o = 128; o > 0; o >>= 1) { if (t < o) red[t] += red[t + o]; __syncthreads(); }
  const float inv = rsqrtf(red[0] / (float)D_ + EPS_);
  float* ob = OUT + (size_t)b * D_ * S_ + s;
  for (int d = t; d < D_; d += 256) ob[(size_t)d * S_] = xb[(size_t)d * S_] * inv * Wn[d];
}

// ---------- Generic GEMM: OUT[b,e,s] = sum_d H[b,d,s] * W[d,e] (+RES) ----------
__global__ __launch_bounds__(256)
void gemm_kernel(const float* __restrict__ H, const float* __restrict__ W,
                 float* __restrict__ OUT, const float* __restrict__ RES,
                 int Dk, int E, int addRes) {
  __shared__ unsigned int ldsA[128 * 16];  // [e_local][k/2]  (weights^T, bf16 pairs)
  __shared__ unsigned int ldsB[128 * 16];  // [s][k/2]        (activations)
  const int e0   = blockIdx.x * 128;
  const int b    = blockIdx.y;
  const int t    = threadIdx.x;
  const int lane = t & 31;
  const int wave = t >> 5;
  const int m    = t & 127;     // row/col handled by this thread in staging
  const int half = t >> 7;      // which 16-wide K chunk
  const float* Hb = H + (size_t)b * Dk * S_;

  v8f acc[8];
#pragma unroll
  for (int j = 0; j < 8; ++j)
#pragma unroll
    for (int r = 0; r < 8; ++r) acc[j][r] = 0.f;

  for (int k0 = 0; k0 < Dk; k0 += 32) {
    const float* wp = W + (size_t)(k0 + half * 16) * E + e0 + m;
    const float* hp = Hb + (size_t)(k0 + half * 16) * S_ + m;
    if (k0 + 32 < Dk) __builtin_prefetch(wp + (size_t)32 * E, 0, 1);
#pragma unroll
    for (int i = 0; i < 8; ++i) {
      float f0 = wp[(size_t)(2 * i) * E];
      float f1 = wp[(size_t)(2 * i + 1) * E];
      ldsA[m * 16 + half * 8 + i] = pack2bf(f0, f1);
    }
#pragma unroll
    for (int i = 0; i < 8; ++i) {
      float f0 = hp[(size_t)(2 * i) * S_];
      float f1 = hp[(size_t)(2 * i + 1) * S_];
      ldsB[m * 16 + half * 8 + i] = pack2bf(f0, f1);
    }
    __syncthreads();
    v16bf a = load_frag_a(ldsA, wave * 16 + (lane & 15), lane);
#pragma unroll
    for (int j = 0; j < 8; ++j) {
      v16bf bf = load_frag_b(ldsB, j * 16 + (lane & 15), lane);
      acc[j] = __builtin_amdgcn_wmma_f32_16x16x32_bf16(false, a, false, bf,
                                                       (short)0, acc[j], false, false);
    }
    __syncthreads();
  }

  const int sc = lane & 15;
  const int hi = lane >> 4;
#pragma unroll
  for (int j = 0; j < 8; ++j)
#pragma unroll
    for (int r = 0; r < 8; ++r) {
      const int e = e0 + wave * 16 + r + 8 * hi;
      const int s = j * 16 + sc;
      const size_t idx = ((size_t)b * E + e) * S_ + s;
      float v = acc[j][r];
      if (addRes) v += RES[idx];
      OUT[idx] = v;
    }
}

// ---------- RoPE in place on (B,NH,HD,S) ----------
__global__ __launch_bounds__(256)
void rope_kernel(float* __restrict__ buf, const int* __restrict__ positions, int NH) {
  const int idx = blockIdx.x * 256 + threadIdx.x;
  const int total = B_ * NH * (HD_ / 2) * S_;
  if (idx >= total) return;
  const int s = idx % S_;
  int tmp = idx / S_;
  const int f = tmp % (HD_ / 2); tmp /= (HD_ / 2);
  const int h = tmp % NH;
  const int b = tmp / NH;
  const float inv = __expf(-logf(10000.f) * (2.0f * (float)f / (float)HD_));
  const float ang = (float)positions[b * S_ + s] * inv;
  float sn, cs; __sincosf(ang, &sn, &cs);
  const size_t i1 = (((size_t)(b * NH + h)) * HD_ + f) * S_ + s;
  const size_t i2 = i1 + (size_t)(HD_ / 2) * S_;
  const float x1 = buf[i1], x2 = buf[i2];
  buf[i1] = x1 * cs - x2 * sn;
  buf[i2] = x2 * cs + x1 * sn;
}

// ---------- transpose fresh K: (B,HKV,HD,S) -> (B,HKV,S,HD) ----------
__global__ __launch_bounds__(256)
void transpose_k_kernel(const float* __restrict__ K, float* __restrict__ KT) {
  const int idx = blockIdx.x * 256 + threadIdx.x;
  const int total = B_ * HKV_ * HD_ * S_;
  if (idx >= total) return;
  const int s = idx % S_;
  int tmp = idx / S_;
  const int d = tmp % HD_;
  const int bk = tmp / HD_;
  KT[((size_t)bk * S_ + s) * HD_ + d] = K[idx];
}

// ---------- self-attn scores: P[b,h,s,a] = scale * sum_d q[d,s]*key[a,d] + mask ----------
__global__ __launch_bounds__(256)
void sa_score_kernel(const float* __restrict__ Q,   // (B,HQ,HD,S)
                     const float* __restrict__ KT,  // fresh (B,HKV,S,HD)
                     const float* __restrict__ KC,  // (L*B,HKV,A,HD)
                     const float* __restrict__ MASK,// (B,1,S,A)
                     const int* __restrict__ WIDX,
                     float* __restrict__ P, int layer) {
  __shared__ unsigned int ldsA[128 * 16];  // [s][d/2]
  __shared__ unsigned int ldsB[128 * 16];  // [a_local][d/2]
  const int a0 = blockIdx.x * 128;
  const int bh = blockIdx.y;
  const int b  = bh / HQ_, h = bh % HQ_;
  const int kv = h / G_;
  const int w0 = WIDX[0];
  const int t = threadIdx.x, lane = t & 31, wave = t >> 5;
  const int m = t & 127, half = t >> 7;
  const float* qh = Q  + ((size_t)(b * HQ_ + h)) * HD_ * S_;
  const float* kf = KT + ((size_t)(b * HKV_ + kv)) * S_ * HD_;
  const float* kc = KC + ((size_t)((layer * B_ + b) * HKV_ + kv)) * A_ * HD_;

  v8f acc[8];
#pragma unroll
  for (int j = 0; j < 8; ++j)
#pragma unroll
    for (int r = 0; r < 8; ++r) acc[j][r] = 0.f;

  const int ag = a0 + m;
  const float* krow = (ag >= w0 && ag < w0 + S_) ? (kf + (size_t)(ag - w0) * HD_)
                                                 : (kc + (size_t)ag * HD_);
  for (int k0 = 0; k0 < HD_; k0 += 32) {
#pragma unroll
    for (int i = 0; i < 8; ++i) {
      float f0 = qh[(size_t)(k0 + half * 16 + 2 * i) * S_ + m];
      float f1 = qh[(size_t)(k0 + half * 16 + 2 * i + 1) * S_ + m];
      ldsA[m * 16 + half * 8 + i] = pack2bf(f0, f1);
    }
#pragma unroll
    for (int i = 0; i < 8; ++i) {
      float f0 = krow[k0 + half * 16 + 2 * i];
      float f1 = krow[k0 + half * 16 + 2 * i + 1];
      ldsB[m * 16 + half * 8 + i] = pack2bf(f0, f1);
    }
    __syncthreads();
    v16bf a = load_frag_a(ldsA, wave * 16 + (lane & 15), lane);
#pragma unroll
    for (int j = 0; j < 8; ++j) {
      v16bf bf = load_frag_b(ldsB, j * 16 + (lane & 15), lane);
      acc[j] = __builtin_amdgcn_wmma_f32_16x16x32_bf16(false, a, false, bf,
                                                       (short)0, acc[j], false, false);
    }
    __syncthreads();
  }

  const int sc = lane & 15, hi = lane >> 4;
#pragma unroll
  for (int j = 0; j < 8; ++j)
#pragma unroll
    for (int r = 0; r < 8; ++r) {
      const int s = wave * 16 + r + 8 * hi;
      const int a = a0 + j * 16 + sc;
      const float msk = MASK[((size_t)b * S_ + s) * A_ + a];
      P[(((size_t)(b * HQ_ + h)) * S_ + s) * A_ + a] = acc[j][r] * SCALE_ + msk;
    }
}

// ---------- row softmax, row length = len ----------
__global__ __launch_bounds__(256)
void softmax_kernel(float* __restrict__ P, int len) {
  float* row = P + (size_t)blockIdx.x * len;
  __shared__ float red[256];
  const int t = threadIdx.x;
  float mx = -3.4e38f;
  for (int i = t; i < len; i += 256) mx = fmaxf(mx, row[i]);
  red[t] = mx; __syncthreads();
  for (int o = 128; o > 0; o >>= 1) { if (t < o) red[t] = fmaxf(red[t], red[t + o]); __syncthreads(); }
  mx = red[0]; __syncthreads();
  float sum = 0.f;
  for (int i = t; i < len; i += 256) { float e = __expf(row[i] - mx); row[i] = e; sum += e; }
  red[t] = sum; __syncthreads();
  for (int o = 128; o > 0; o >>= 1) { if (t < o) red[t] += red[t + o]; __syncthreads(); }
  const float inv = 1.0f / red[0];
  __syncthreads();
  for (int i = t; i < len; i += 256) row[i] *= inv;
}

// ---------- self attn*V: OUT[b,h*HD+d,s] = sum_a val[d,a]*P[s,a] ----------
__global__ __launch_bounds__(256)
void sa_attnv_kernel(const float* __restrict__ P,   // (B,HQ,S,A)
                     const float* __restrict__ VF,  // fresh (B,HKV,HD,S)
                     const float* __restrict__ VC,  // (L*B,HKV,HD,A)
                     const int* __restrict__ WIDX,
                     float* __restrict__ OUT, int layer) {
  __shared__ unsigned int ldsA[128 * 16];  // [d][a/2]
  __shared__ unsigned int ldsB[128 * 16];  // [s][a/2]
  const int bh = blockIdx.x;
  const int b = bh / HQ_, h = bh % HQ_;
  const int kv = h / G_;
  const int w0 = WIDX[0];
  const int t = threadIdx.x, lane = t & 31, wave = t >> 5;
  const int m = t & 127, half = t >> 7;
  const float* vcrow = VC + (((size_t)((layer * B_ + b) * HKV_ + kv)) * HD_ + m) * A_;
  const float* vfrow = VF + (((size_t)(b * HKV_ + kv)) * HD_ + m) * S_;
  const float* prow  = P  + (((size_t)(b * HQ_ + h)) * S_ + m) * A_;

  v8f acc[8];
#pragma unroll
  for (int j = 0; j < 8; ++j)
#pragma unroll
    for (int r = 0; r < 8; ++r) acc[j][r] = 0.f;

  for (int k0 = 0; k0 < A_; k0 += 32) {
#pragma unroll
    for (int i = 0; i < 8; ++i) {
      const int a0 = k0 + half * 16 + 2 * i;
      const int a1 = a0 + 1;
      float f0 = (a0 >= w0 && a0 < w0 + S_) ? vfrow[a0 - w0] : vcrow[a0];
      float f1 = (a1 >= w0 && a1 < w0 + S_) ? vfrow[a1 - w0] : vcrow[a1];
      ldsA[m * 16 + half * 8 + i] = pack2bf(f0, f1);
    }
#pragma unroll
    for (int i = 0; i < 8; ++i) {
      float f0 = prow[k0 + half * 16 + 2 * i];
      float f1 = prow[k0 + half * 16 + 2 * i + 1];
      ldsB[m * 16 + half * 8 + i] = pack2bf(f0, f1);
    }
    __syncthreads();
    v16bf a = load_frag_a(ldsA, wave * 16 + (lane & 15), lane);
#pragma unroll
    for (int j = 0; j < 8; ++j) {
      v16bf bf = load_frag_b(ldsB, j * 16 + (lane & 15), lane);
      acc[j] = __builtin_amdgcn_wmma_f32_16x16x32_bf16(false, a, false, bf,
                                                       (short)0, acc[j], false, false);
    }
    __syncthreads();
  }

  const int sc = lane & 15, hi = lane >> 4;
#pragma unroll
  for (int j = 0; j < 8; ++j)
#pragma unroll
    for (int r = 0; r < 8; ++r) {
      const int d = wave * 16 + r + 8 * hi;
      const int s = j * 16 + sc;
      OUT[(((size_t)(b * HQ_ + h)) * HD_ + d) * S_ + s] = acc[j][r];
    }
}

// ---------- cross-attn scores: Pc[b,h,s,t] ----------
__global__ __launch_bounds__(256)
void ca_score_kernel(const float* __restrict__ Q,   // (B,HC,HDC,S)
                     const float* __restrict__ CK,  // (L*B,HC,T,HDC)
                     const int* __restrict__ ENC,
                     float* __restrict__ Pc, int layer) {
  __shared__ unsigned int ldsA[128 * 16];
  __shared__ unsigned int ldsB[128 * 16];
  const int t0 = blockIdx.x * 128;
  const int bh = blockIdx.y;
  const int b = bh / HC_, h = bh % HC_;
  const int encl = ENC[b];
  const int t = threadIdx.x, lane = t & 31, wave = t >> 5;
  const int m = t & 127, half = t >> 7;
  const float* qh = Q  + ((size_t)(b * HC_ + h)) * HDC_ * S_;
  const float* kr = CK + (((size_t)((layer * B_ + b) * HC_ + h)) * T_ + (t0 + m)) * HDC_;

  v8f acc[8];
#pragma unroll
  for (int j = 0; j < 8; ++j)
#pragma unroll
    for (int r = 0; r < 8; ++r) acc[j][r] = 0.f;

  for (int k0 = 0; k0 < HDC_; k0 += 32) {
#pragma unroll
    for (int i = 0; i < 8; ++i) {
      float f0 = qh[(size_t)(k0 + half * 16 + 2 * i) * S_ + m];
      float f1 = qh[(size_t)(k0 + half * 16 + 2 * i + 1) * S_ + m];
      ldsA[m * 16 + half * 8 + i] = pack2bf(f0, f1);
    }
#pragma unroll
    for (int i = 0; i < 8; ++i) {
      float f0 = kr[k0 + half * 16 + 2 * i];
      float f1 = kr[k0 + half * 16 + 2 * i + 1];
      ldsB[m * 16 + half * 8 + i] = pack2bf(f0, f1);
    }
    __syncthreads();
    v16bf a = load_frag_a(ldsA, wave * 16 + (lane & 15), lane);
#pragma unroll
    for (int j = 0; j < 8; ++j) {
      v16bf bf = load_frag_b(ldsB, j * 16 + (lane & 15), lane);
      acc[j] = __builtin_amdgcn_wmma_f32_16x16x32_bf16(false, a, false, bf,
                                                       (short)0, acc[j], false, false);
    }
    __syncthreads();
  }

  const int sc = lane & 15, hi = lane >> 4;
#pragma unroll
  for (int j = 0; j < 8; ++j)
#pragma unroll
    for (int r = 0; r < 8; ++r) {
      const int s  = wave * 16 + r + 8 * hi;
      const int tt = t0 + j * 16 + sc;
      const float msk = (tt < encl) ? 0.f : NEG_;
      Pc[(((size_t)(b * HC_ + h)) * S_ + s) * T_ + tt] = acc[j][r] * SCALE_ + msk;
    }
}

// ---------- cross attn*V ----------
__global__ __launch_bounds__(256)
void ca_attnv_kernel(const float* __restrict__ Pc,  // (B,HC,S,T)
                     const float* __restrict__ CV,  // (L*B,HC,HDC,T)
                     float* __restrict__ OUT, int layer) {
  __shared__ unsigned int ldsA[128 * 16];
  __shared__ unsigned int ldsB[128 * 16];
  const int bh = blockIdx.x;
  const int b = bh / HC_, h = bh % HC_;
  const int t = threadIdx.x, lane = t & 31, wave = t >> 5;
  const int m = t & 127, half = t >> 7;
  const float* vrow = CV + (((size_t)((layer * B_ + b) * HC_ + h)) * HDC_ + m) * T_;
  const float* prow = Pc + (((size_t)(b * HC_ + h)) * S_ + m) * T_;

  v8f acc[8];
#pragma unroll
  for (int j = 0; j < 8; ++j)
#pragma unroll
    for (int r = 0; r < 8; ++r) acc[j][r] = 0.f;

  for (int k0 = 0; k0 < T_; k0 += 32) {
#pragma unroll
    for (int i = 0; i < 8; ++i)
      ldsA[m * 16 + half * 8 + i] = pack2bf(vrow[k0 + half * 16 + 2 * i],
                                            vrow[k0 + half * 16 + 2 * i + 1]);
#pragma unroll
    for (int i = 0; i < 8; ++i)
      ldsB[m * 16 + half * 8 + i] = pack2bf(prow[k0 + half * 16 + 2 * i],
                                            prow[k0 + half * 16 + 2 * i + 1]);
    __syncthreads();
    v16bf a = load_frag_a(ldsA, wave * 16 + (lane & 15), lane);
#pragma unroll
    for (int j = 0; j < 8; ++j) {
      v16bf bf = load_frag_b(ldsB, j * 16 + (lane & 15), lane);
      acc[j] = __builtin_amdgcn_wmma_f32_16x16x32_bf16(false, a, false, bf,
                                                       (short)0, acc[j], false, false);
    }
    __syncthreads();
  }

  const int sc = lane & 15, hi = lane >> 4;
#pragma unroll
  for (int j = 0; j < 8; ++j)
#pragma unroll
    for (int r = 0; r < 8; ++r) {
      const int d = wave * 16 + r + 8 * hi;
      const int s = j * 16 + sc;
      OUT[(((size_t)(b * HC_ + h)) * HDC_ + d) * S_ + s] = acc[j][r];
    }
}

// ---------- silu(gate)*up -> gate ----------
__global__ __launch_bounds__(256)
void silu_mul_kernel(float* __restrict__ gate, const float* __restrict__ up) {
  const size_t idx = (size_t)blockIdx.x * 256 + threadIdx.x;
  const size_t total = (size_t)B_ * FF_ * S_;
  if (idx >= total) return;
  const float g = gate[idx];
  gate[idx] = (g / (1.0f + __expf(-g))) * up[idx];
}

// ---------- host orchestration ----------
extern "C" void kernel_launch(void* const* d_in, const int* in_sizes, int n_in,
                              void* d_out, int out_size, void* d_ws, size_t ws_size,
                              hipStream_t stream) {
  const float* x_in      = (const float*)d_in[0];
  const int*   positions = (const int*)  d_in[1];
  const int*   widx      = (const int*)  d_in[2];
  const float* samask    = (const float*)d_in[3];
  const int*   enclen    = (const int*)  d_in[4];
  const float* q_w  = (const float*)d_in[5];
  const float* k_w  = (const float*)d_in[6];
  const float* v_w  = (const float*)d_in[7];
  const float* o_w  = (const float*)d_in[8];
  const float* cq_w = (const float*)d_in[9];
  const float* co_w = (const float*)d_in[10];
  const float* sa_n = (const float*)d_in[11];
  const float* ca_n = (const float*)d_in[12];
  const float* ml_n = (const float*)d_in[13];
  const float* fi_n = (const float*)d_in[14];
  const float* wg_w = (const float*)d_in[15];
  const float* wu_w = (const float*)d_in[16];
  const float* wd_w = (const float*)d_in[17];
  const float* kc   = (const float*)d_in[18];
  const float* vc   = (const float*)d_in[19];
  const float* ckc  = (const float*)d_in[20];
  const float* cvc  = (const float*)d_in[21];

  float* ws = (float*)d_ws;
  float* x    = ws; ws += (size_t)B_ * D_ * S_;
  float* h    = ws; ws += (size_t)B_ * D_ * S_;
  float* q    = ws; ws += (size_t)B_ * HQ_ * HD_ * S_;
  float* kbuf = ws; ws += (size_t)B_ * HKV_ * HD_ * S_;
  float* kT   = ws; ws += (size_t)B_ * HKV_ * S_ * HD_;
  float* vbuf = ws; ws += (size_t)B_ * HKV_ * HD_ * S_;
  float* attn = ws; ws += (size_t)B_ * HQ_ * HD_ * S_;
  float* P    = ws; ws += (size_t)B_ * HQ_ * S_ * A_;
  float* Pc   = ws; ws += (size_t)B_ * HC_ * S_ * T_;
  float* gate = ws; ws += (size_t)B_ * FF_ * S_;
  float* up   = ws; ws += (size_t)B_ * FF_ * S_;

  hipMemcpyAsync(x, x_in, sizeof(float) * (size_t)B_ * D_ * S_,
                 hipMemcpyDeviceToDevice, stream);

  const dim3 blk(256);
  for (int l = 0; l < L_; ++l) {
    const float* qw  = q_w  + (size_t)l * D_ * (HQ_ * HD_);
    const float* kw  = k_w  + (size_t)l * D_ * (HKV_ * HD_);
    const float* vw  = v_w  + (size_t)l * D_ * (HKV_ * HD_);
    const float* ow  = o_w  + (size_t)l * (HQ_ * HD_) * D_;
    const float* cqw = cq_w + (size_t)l * D_ * (HC_ * HDC_);
    const float* cow = co_w + (size_t)l * (HC_ * HDC_) * D_;
    const float* wgw = wg_w + (size_t)l * D_ * FF_;
    const float* wuw = wu_w + (size_t)l * D_ * FF_;
    const float* wdw = wd_w + (size_t)l * FF_ * D_;

    // --- self attention ---
    rmsnorm_kernel<<<B_ * S_, blk, 0, stream>>>(x, sa_n + (size_t)l * D_, h);
    gemm_kernel<<<dim3((HQ_ * HD_) / 128, B_), blk, 0, stream>>>(h, qw, q, nullptr, D_, HQ_ * HD_, 0);
    gemm_kernel<<<dim3((HKV_ * HD_) / 128, B_), blk, 0, stream>>>(h, kw, kbuf, nullptr, D_, HKV_ * HD_, 0);
    gemm_kernel<<<dim3((HKV_ * HD_) / 128, B_), blk, 0, stream>>>(h, vw, vbuf, nullptr, D_, HKV_ * HD_, 0);
    rope_kernel<<<(B_ * HQ_ * (HD_ / 2) * S_ + 255) / 256, blk, 0, stream>>>(q, positions, HQ_);
    rope_kernel<<<(B_ * HKV_ * (HD_ / 2) * S_ + 255) / 256, blk, 0, stream>>>(kbuf, positions, HKV_);
    transpose_k_kernel<<<(B_ * HKV_ * HD_ * S_ + 255) / 256, blk, 0, stream>>>(kbuf, kT);
    sa_score_kernel<<<dim3(A_ / 128, B_ * HQ_), blk, 0, stream>>>(q, kT, kc, samask, widx, P, l);
    softmax_kernel<<<B_ * HQ_ * S_, blk, 0, stream>>>(P, A_);
    sa_attnv_kernel<<<B_ * HQ_, blk, 0, stream>>>(P, vbuf, vc, widx, attn, l);
    gemm_kernel<<<dim3(D_ / 128, B_), blk, 0, stream>>>(attn, ow, x, x, HQ_ * HD_, D_, 1);

    // --- cross attention ---
    rmsnorm_kernel<<<B_ * S_, blk, 0, stream>>>(x, ca_n + (size_t)l * D_, h);
    gemm_kernel<<<dim3((HC_ * HDC_) / 128, B_), blk, 0, stream>>>(h, cqw, q, nullptr, D_, HC_ * HDC_, 0);
    rope_kernel<<<(B_ * HC_ * (HDC_ / 2) * S_ + 255) / 256, blk, 0, stream>>>(q, positions, HC_);
    ca_score_kernel<<<dim3(T_ / 128, B_ * HC_), blk, 0, stream>>>(q, ckc, enclen, Pc, l);
    softmax_kernel<<<B_ * HC_ * S_, blk, 0, stream>>>(Pc, T_);
    ca_attnv_kernel<<<B_ * HC_, blk, 0, stream>>>(Pc, cvc, attn, l);
    gemm_kernel<<<dim3(D_ / 128, B_), blk, 0, stream>>>(attn, cow, x, x, HC_ * HDC_, D_, 1);

    // --- MLP ---
    rmsnorm_kernel<<<B_ * S_, blk, 0, stream>>>(x, ml_n + (size_t)l * D_, h);
    gemm_kernel<<<dim3(FF_ / 128, B_), blk, 0, stream>>>(h, wgw, gate, nullptr, D_, FF_, 0);
    gemm_kernel<<<dim3(FF_ / 128, B_), blk, 0, stream>>>(h, wuw, up, nullptr, D_, FF_, 0);
    silu_mul_kernel<<<(int)(((size_t)B_ * FF_ * S_ + 255) / 256), blk, 0, stream>>>(gate, up);
    gemm_kernel<<<dim3(D_ / 128, B_), blk, 0, stream>>>(gate, wdw, x, x, FF_, D_, 1);
  }

  rmsnorm_kernel<<<B_ * S_, blk, 0, stream>>>(x, fi_n, (float*)d_out);
}
